// SimilarityTreeLSTM_29154238005603
// MI455X (gfx1250) — matrix operand order repaired and checked by reading, not compiled, not run on hardware
//
#include <hip/hip_runtime.h>
#include <cstdint>
#include <cstddef>

// ---------------------------------------------------------------------------
// Types for CDNA5 WMMA (gfx1250, wave32)
// ---------------------------------------------------------------------------
typedef __attribute__((ext_vector_type(16))) __bf16 v16bf;
typedef __attribute__((ext_vector_type(8)))  __bf16 v8bf;
typedef __attribute__((ext_vector_type(8)))  float  v8f;

#define S_LEN 1024
#define DIM   512
#define GATES 2048   // 3M (iou) + M (f)
#define IOU   1536

static __device__ __forceinline__ float sigf(float x) {
    return 1.0f / (1.0f + __expf(-x));
}

static __device__ __forceinline__ v16bf cat8(v8bf lo, v8bf hi) {
    return __builtin_shufflevector(lo, hi, 0,1,2,3,4,5,6,7,8,9,10,11,12,13,14,15);
}

// gfx1250 async global->LDS copy (ASYNCcnt-tracked), GV addressing form.
// lds_off: 32-bit LDS byte address (low 32 bits of a generic pointer to LDS).
static __device__ __forceinline__ void async_copy_b128(unsigned lds_off, const void* gaddr) {
    asm volatile("global_load_async_to_lds_b128 %0, %1, off"
                 :: "v"(lds_off), "v"(gaddr) : "memory");
}
static __device__ __forceinline__ void wait_async0() {
    asm volatile("s_wait_asynccnt 0x0" ::: "memory");
}

// ---------------------------------------------------------------------------
// Kernel 1: gather embedding rows for both sequences, convert to bf16
//   grid (1024, 2) x 256
// ---------------------------------------------------------------------------
__global__ void k_gather(const int* __restrict__ l_ids, const int* __restrict__ r_ids,
                         const float* __restrict__ emb, __bf16* __restrict__ Xbf) {
    const int t   = blockIdx.x;
    const int seq = blockIdx.y;
    const int id  = (seq == 0 ? l_ids : r_ids)[t];
    const float* src = emb + (size_t)id * DIM;
    __bf16* dst = Xbf + ((size_t)seq * S_LEN + t) * DIM;
    for (int k = threadIdx.x; k < DIM; k += blockDim.x)
        dst[k] = (__bf16)src[k];
}

// ---------------------------------------------------------------------------
// Kernel 2: convert weights to bf16 (combined [2048 x 512] row blocks:
//   rows [0,1536) = iou weights, rows [1536,2048) = f weights) + fused bias
//   grid 2048 x 256
// ---------------------------------------------------------------------------
__global__ void k_wconv(const float* __restrict__ Wioux, const float* __restrict__ Wfx,
                        const float* __restrict__ Wiouh, const float* __restrict__ Wfh,
                        const float* __restrict__ bioux, const float* __restrict__ biouh,
                        const float* __restrict__ bfx,   const float* __restrict__ bfh,
                        __bf16* __restrict__ Wx, __bf16* __restrict__ Wrec,
                        float* __restrict__ biasc) {
    const int row = blockIdx.x;
    const float* sx = (row < IOU) ? (Wioux + (size_t)row * DIM) : (Wfx + (size_t)(row - IOU) * DIM);
    const float* sr = (row < IOU) ? (Wiouh + (size_t)row * DIM) : (Wfh + (size_t)(row - IOU) * DIM);
    for (int k = threadIdx.x; k < DIM; k += blockDim.x) {
        Wx  [(size_t)row * DIM + k] = (__bf16)sx[k];
        Wrec[(size_t)row * DIM + k] = (__bf16)sr[k];
    }
    if (threadIdx.x == 0)
        biasc[row] = (row < IOU) ? (bioux[row] + biouh[row]) : (bfx[row - IOU] + bfh[row - IOU]);
}

// ---------------------------------------------------------------------------
// Kernel 3: WMMA bf16 GEMM   pre[s][t][j] = sum_k X[s][t][k] * Wx[j][k] + biasc[j]
//   pre: [2][1024][2048] f32.  grid (64 Mtiles, 16 Nsupers, 2 seq) x 256 (8 waves)
// ---------------------------------------------------------------------------
__global__ void k_gemm_pre(const __bf16* __restrict__ Xbf, const __bf16* __restrict__ Wx,
                           const float* __restrict__ biasc, float* __restrict__ pre) {
    const int lane = threadIdx.x & 31;
    const int wave = threadIdx.x >> 5;
    const int m0   = blockIdx.x * 16;
    const int n0   = blockIdx.y * 128 + wave * 16;
    const int seq  = blockIdx.z;

    const __bf16* Xp = Xbf + (size_t)seq * S_LEN * DIM;
    const int mrow    = m0 + (lane & 15);
    const int khalf8  = (lane >> 4) * 8;
    const int khalf16 = (lane >> 4) * 16;
    const int ncol    = n0 + (lane & 15);

    v8f acc = {};
#pragma unroll
    for (int k0 = 0; k0 < DIM; k0 += 32) {
        // A: 16x32 bf16 tile of X (row striped within lane, contiguous k = 16h + s)
        const __bf16* ap = Xp + (size_t)mrow * DIM + k0 + khalf16;
        v16bf a = cat8(*(const v8bf*)ap, *(const v8bf*)(ap + 8));
        // B: 32x16 tile; B[k][n] = Wx[n][k] -> contiguous row-major read
        v16bf b = *(const v16bf*)(Wx + (size_t)ncol * DIM + k0 + khalf16);
        acc = __builtin_amdgcn_wmma_f32_16x16x32_bf16(false, a, false, b,
                                                      (short)0, acc, false, false);
    }

    const float bsum = biasc[ncol];
    float* outp = pre + (size_t)seq * S_LEN * GATES;
    const int rbase = (lane >> 4) * 8;
#pragma unroll
    for (int r = 0; r < 8; ++r)
        outp[(size_t)(m0 + rbase + r) * GATES + ncol] = acc[r] + bsum;
}

// ---------------------------------------------------------------------------
// Kernel 4: sequential recurrence, both sequences in the WMMA N-dimension.
//   1 block x 512 threads (16 waves); each wave owns 128 gate rows (8 tiles).
//   gates[m][n] = sum_k Wrec[m][k] * H[k][n],  H columns 0/1 = h_l/h_r,
//   columns 2..15 read an always-zero LDS row (EXEC stays all-ones).
//   Next step's pre[] activations are double-buffered into LDS with
//   global_load_async_to_lds_b128 overlapped with the WMMA phase.
// ---------------------------------------------------------------------------
__global__ void __launch_bounds__(512)
k_recur(const __bf16* __restrict__ Wrec, const float* __restrict__ pre,
        float* __restrict__ hF) {
    __shared__ __align__(32) __bf16 hbf[4][DIM];      //  4 KiB (rows 2..3 stay zero)
    __shared__ __align__(16) float  gates[2][GATES];  // 16 KiB
    __shared__ __align__(16) float  cst[2][DIM];      //  4 KiB
    __shared__ __align__(16) float  preLds[2][2 * GATES]; // 32 KiB double buffer

    const int tid  = threadIdx.x;
    const int lane = tid & 31;
    const int wave = tid >> 5;
    const int mbase   = wave * 128;
    const int mlane   = lane & 15;
    const int hrow    = (mlane < 2) ? mlane : 2;   // zero column source for n>=2
    const int khalf8  = (lane >> 4) * 8;
    const int khalf16 = (lane >> 4) * 16;

    // LDS byte addresses for the async destination (low 32 bits of generic ptr)
    const unsigned preLdsBase = (unsigned)(size_t)&preLds[0][0];
    // this thread's two b128 chunks: flat f32 index f = tid*8 + p*4 over [0,4096)
    const int f0 = tid * 8;
    const int seqA = f0 >> 11,        jA = f0 & (GATES - 1);
    const int seqB = (f0 + 4) >> 11,  jB = (f0 + 4) & (GATES - 1);

    auto prefetch_step = [&](int s, int buf) {
        const float* ga = pre + ((size_t)(seqA * S_LEN + s) * GATES + jA);
        const float* gb = pre + ((size_t)(seqB * S_LEN + s) * GATES + jB);
        const unsigned l0 = preLdsBase + (unsigned)buf * (2 * GATES * 4) + (unsigned)f0 * 4;
        async_copy_b128(l0,      ga);
        async_copy_b128(l0 + 16, gb);
    };

    // zero-init state
    for (int i = tid; i < 4 * DIM; i += 512)
        ((__bf16*)hbf)[i] = (__bf16)0.0f;
    for (int i = tid; i < 2 * DIM; i += 512)
        ((float*)cst)[i] = 0.0f;

    prefetch_step(0, 0);   // warm the pipeline
    __syncthreads();

    for (int step = 0; step < S_LEN; ++step) {
        v8f acc[8];
#pragma unroll
        for (int t = 0; t < 8; ++t) acc[t] = (v8f){};

#pragma unroll
        for (int k0i = 0; k0i < 16; ++k0i) {
            const int k0 = k0i * 32;
            // B: H tile 32x16 from LDS (column = lane&15; zeros for n>=2)
            v16bf b = *(const v16bf*)&hbf[hrow][k0 + khalf16];
#pragma unroll
            for (int t = 0; t < 8; ++t) {
                const __bf16* ap = Wrec + (size_t)(mbase + t * 16 + mlane) * DIM + k0 + khalf16;
                v16bf a = cat8(*(const v8bf*)ap, *(const v8bf*)(ap + 8));
                acc[t] = __builtin_amdgcn_wmma_f32_16x16x32_bf16(false, a, false, b,
                                                                 (short)0, acc[t], false, false);
            }
        }

        // scatter valid columns (n==0,1) of D into gates[seq][row]
        if (mlane < 2) {
            const int seq   = mlane;
            const int rbase = (lane >> 4) * 8;
#pragma unroll
            for (int t = 0; t < 8; ++t)
#pragma unroll
                for (int r = 0; r < 8; ++r)
                    gates[seq][mbase + t * 16 + rbase + r] = acc[t][r];
        }

        wait_async0();        // this thread's pre[] prefetch landed in LDS
        __syncthreads();      // ... and everyone else's too

        // pointwise LSTM cell (f32), one j per thread per sequence
        {
            const int j = tid;
            const float* P = &preLds[step & 1][0];
#pragma unroll
            for (int s = 0; s < 2; ++s) {
                const int b = s * GATES;
                const float gi = gates[s][j]        + P[b + j];
                const float go = gates[s][512 + j]  + P[b + 512 + j];
                const float gu = gates[s][1024 + j] + P[b + 1024 + j];
                const float gf = gates[s][1536 + j] + P[b + 1536 + j];
                const float c  = sigf(gi) * tanhf(gu) + sigf(gf) * cst[s][j];
                cst[s][j] = c;
                const float h = sigf(go) * tanhf(c);
                hbf[s][j] = (__bf16)h;
                if (step == S_LEN - 1) hF[s * DIM + j] = h;
            }
        }

        // prefetch next step's pre[] into the other buffer (overlaps next WMMA
        // phase; that buffer was last read two barriers ago -> WAR-safe)
        if (step + 1 < S_LEN)
            prefetch_step(step + 1, (step + 1) & 1);

        __syncthreads();
    }
}

// ---------------------------------------------------------------------------
// Kernel 5: head.  hB from pre[s][1023] (zero-state cell), similarity vector,
//   Wh matvec + sigmoid, Wp matvec, log_softmax.  1 block x 256.
// ---------------------------------------------------------------------------
__global__ void k_head(const float* __restrict__ hF, const float* __restrict__ pre,
                       const float* __restrict__ Wh, const float* __restrict__ bh,
                       const float* __restrict__ Wp, const float* __restrict__ bp,
                       float* __restrict__ out) {
    __shared__ float vec[GATES];   // [ hF_l*hF_r | hB_l*hB_r | |hF_l-hF_r| | |hB_l-hB_r| ]
    __shared__ float hid[256];

    const int tid = threadIdx.x;
    const float* pl = pre + (size_t)(0 * S_LEN + (S_LEN - 1)) * GATES;
    const float* pr = pre + (size_t)(1 * S_LEN + (S_LEN - 1)) * GATES;

    for (int j = tid; j < DIM; j += blockDim.x) {
        const float hFl = hF[j], hFr = hF[DIM + j];
        // backward vector: cell(x_last, c=0, h=0) -> c2=sig(i)tanh(u), h2=sig(o)tanh(c2)
        const float cBl = sigf(pl[j]) * tanhf(pl[1024 + j]);
        const float hBl = sigf(pl[512 + j]) * tanhf(cBl);
        const float cBr = sigf(pr[j]) * tanhf(pr[1024 + j]);
        const float hBr = sigf(pr[512 + j]) * tanhf(cBr);
        vec[j]          = hFl * hFr;
        vec[512  + j]   = hBl * hBr;
        vec[1024 + j]   = fabsf(hFl - hFr);
        vec[1536 + j]   = fabsf(hBl - hBr);
    }
    __syncthreads();

    // hidden layer: 256 rows, one per thread
    {
        float acc = bh[tid];
        const float* wr = Wh + (size_t)tid * GATES;
        for (int k = 0; k < GATES; ++k) acc = fmaf(wr[k], vec[k], acc);
        hid[tid] = sigf(acc);
    }
    __syncthreads();

    if (tid == 0) {
        float lg[5];
        float mx = -3.4e38f;
#pragma unroll
        for (int c = 0; c < 5; ++c) {
            float s = bp[c];
            for (int k = 0; k < 256; ++k) s = fmaf(Wp[c * 256 + k], hid[k], s);
            lg[c] = s;
            mx = fmaxf(mx, s);
        }
        float se = 0.0f;
#pragma unroll
        for (int c = 0; c < 5; ++c) se += __expf(lg[c] - mx);
        const float lse = mx + __logf(se);
#pragma unroll
        for (int c = 0; c < 5; ++c) out[c] = lg[c] - lse;
    }
}

// ---------------------------------------------------------------------------
// Launch
// ---------------------------------------------------------------------------
extern "C" void kernel_launch(void* const* d_in, const int* in_sizes, int n_in,
                              void* d_out, int out_size, void* d_ws, size_t ws_size,
                              hipStream_t stream) {
    (void)in_sizes; (void)n_in; (void)out_size; (void)ws_size;

    const int*   l_ids = (const int*)  d_in[0];
    const int*   r_ids = (const int*)  d_in[1];
    const float* emb   = (const float*)d_in[2];
    const float* Wioux = (const float*)d_in[3];
    const float* bioux = (const float*)d_in[4];
    const float* Wiouh = (const float*)d_in[5];
    const float* biouh = (const float*)d_in[6];
    const float* Wfx   = (const float*)d_in[7];
    const float* bfx   = (const float*)d_in[8];
    const float* Wfh   = (const float*)d_in[9];
    const float* bfh   = (const float*)d_in[10];
    const float* Wh    = (const float*)d_in[11];
    const float* bh    = (const float*)d_in[12];
    const float* Wp    = (const float*)d_in[13];
    const float* bp    = (const float*)d_in[14];

    char* ws = (char*)d_ws;
    size_t off = 0;
    auto carve = [&](size_t bytes) -> char* {
        char* p = ws + off;
        off = (off + bytes + 255) & ~(size_t)255;
        return p;
    };
    __bf16* Xbf   = (__bf16*)carve((size_t)2 * S_LEN * DIM * 2);   // 2 MiB
    __bf16* Wx    = (__bf16*)carve((size_t)GATES * DIM * 2);       // 2 MiB
    __bf16* Wrec  = (__bf16*)carve((size_t)GATES * DIM * 2);       // 2 MiB
    float*  biasc = (float*) carve((size_t)GATES * 4);
    float*  pre   = (float*) carve((size_t)2 * S_LEN * GATES * 4); // 16 MiB
    float*  hF    = (float*) carve((size_t)2 * DIM * 4);

    k_gather<<<dim3(S_LEN, 2), 256, 0, stream>>>(l_ids, r_ids, emb, Xbf);
    k_wconv<<<GATES, 256, 0, stream>>>(Wioux, Wfx, Wiouh, Wfh,
                                       bioux, biouh, bfx, bfh, Wx, Wrec, biasc);
    k_gemm_pre<<<dim3(S_LEN / 16, GATES / 128, 2), 256, 0, stream>>>(Xbf, Wx, biasc, pre);
    k_recur<<<1, 512, 0, stream>>>(Wrec, pre, hF);
    k_head<<<1, 256, 0, stream>>>(hF, pre, Wh, bh, Wp, bp, (float*)d_out);
}
